// VNlayer_11630771438303
// MI455X (gfx1250) — matrix operand order repaired and checked by reading, not compiled, not run on hardware
//
#include <hip/hip_runtime.h>
#include <hip/hip_bf16.h>

// ---------------------------------------------------------------------------
// VN edge-conv layer for MI455X (gfx1250, wave32, WMMA + TDM + async-LDS)
//   x:      [B, Cin, 3, N]  f32   (== xf [B, 192, N])
//   W_feat: [Cout, Cin] f32, W_dir: [Cout, Cin] f32, gamma/beta: [Cout]
//   out:    [B, Cout, 3, N] f32
// ---------------------------------------------------------------------------

#define B_    8
#define CIN   64
#define COUT  64
#define NPT   2048
#define DFE   192      // 3*Cin
#define KNB   11       // k+1 neighbors

typedef __attribute__((ext_vector_type(2)))  float  v2f;
typedef __attribute__((ext_vector_type(8)))  float  v8f;
typedef __attribute__((ext_vector_type(16))) __bf16 v16bf;
typedef unsigned int u32x4 __attribute__((ext_vector_type(4)));
typedef int          i32x8 __attribute__((ext_vector_type(8)));
typedef int          i32x4 __attribute__((ext_vector_type(4)));

// low 32 bits of a flat pointer to __shared__ == LDS byte offset (ISA 10.2)
static __device__ __forceinline__ unsigned lds_off(const void* p) {
  return (unsigned)(size_t)p;
}

// ===========================================================================
// Kernel A: xx[b,n] = sum_r xf[b,r,n]^2   (+ zero the BN stats accumulators)
// ===========================================================================
__global__ __launch_bounds__(256) void vn_xx_kernel(const float* __restrict__ x,
                                                    float* __restrict__ xx,
                                                    float* __restrict__ stats) {
  if (blockIdx.x == 0 && threadIdx.x < 128) stats[threadIdx.x] = 0.0f;
  int t = blockIdx.x * 256 + threadIdx.x;        // 0 .. B*N-1
  int b = t >> 11, n = t & (NPT - 1);
  const float* xb = x + (size_t)b * DFE * NPT + n;
  float s = 0.0f;
  for (int r = 0; r < DFE; ++r) {
    float v = xb[(size_t)r * NPT];
    s += v * v;
  }
  xx[t] = s;
}

// ===========================================================================
// Kernel B: fused KNN.  Gram tiles via V_WMMA_F32_16X16X4_F32; the 16-col
// B panel (192 x 16 f32) is DMA'd into LDS by the Tensor Data Mover (one
// tensor_load_to_lds per tile, issued by wave 0, tracked with TENSORcnt).
// Top-11 kept in per-lane sorted register lists (2 lanes per row).
// ===========================================================================
__global__ __launch_bounds__(128) void vn_knn_kernel(const float* __restrict__ x,
                                                     const float* __restrict__ xx,
                                                     int* __restrict__ idx_out) {
  __shared__ float bstage[DFE][16];        // TDM tile (contiguous, bank-clean)
  __shared__ float ctile[4][16][17];       // per-wave score tile
  __shared__ float mval[4][16][22];
  __shared__ int   midx[4][16][22];

  const int tid  = threadIdx.x;
  const int w    = tid >> 5;               // wave id 0..3
  const int lane = tid & 31;
  const int lo   = lane & 15, hi = lane >> 4;
  const int blk  = blockIdx.x;             // 0..255
  const int b    = blk >> 5;               // 8 batches
  const int nb   = ((blk & 31) * 4 + w) * 16;   // this wave's row base
  const float* xf = x + (size_t)b * DFE * NPT;

  // Preload this wave's A fragments for all 48 k-steps (f32 16x16x4 layout:
  // lanes 0-15 hold K=0,1; lanes 16-31 hold K=2,3).
  float a0[48], a1[48];
  {
    const float* ap = xf + (size_t)(2 * hi) * NPT + (nb + lo);
    #pragma unroll
    for (int s = 0; s < 48; ++s) {
      a0[s] = ap[(size_t)(4 * s)     * NPT];
      a1[s] = ap[(size_t)(4 * s + 1) * NPT];
    }
  }

  float lv[11]; int li[11];
  #pragma unroll
  for (int j = 0; j < 11; ++j) { lv[j] = -3.0e38f; li[j] = 0; }

  const unsigned ldsb = lds_off(&bstage[0][0]);

  for (int ct = 0; ct < 128; ++ct) {
    const int mb = ct * 16;
    __syncthreads();                        // previous tile fully consumed
    if (w == 0) {
      // ---- Tensor DMA descriptor (D#): 2D tile 16(x) x 192(y), f32 ----
      unsigned long long ga = (unsigned long long)(size_t)(xf + mb);
      u32x4 g0;
      g0[0] = 1u;                            // count=1, user mode
      g0[1] = ldsb;                          // lds_addr
      g0[2] = (unsigned)(ga & 0xFFFFFFFFu);  // global_addr[31:0]
      g0[3] = (unsigned)((ga >> 32) & 0x01FFFFFFu) | 0x80000000u; // [56:32]|type=2
      i32x8 g1;
      g1[0] = 0x00020000;                    // data_size=4B, no pad/iterate
      g1[1] = (int)(2048u << 16);            // tensor_dim0 = 2048 (bits 79:48)
      g1[2] = (int)(192u  << 16);            // tensor_dim1 = 192  (bits 111:80)
      g1[3] = (int)(16u   << 16);            // tile_dim0   = 16   (bits 127:112)
      g1[4] = 192;                           // tile_dim1=192, tile_dim2=0
      g1[5] = 2048;                          // tensor_dim0_stride[31:0]
      g1[6] = 0;                             // stride hi / dim1_stride
      g1[7] = 0;
      i32x4 z4 = {0, 0, 0, 0};
      i32x8 z8 = {0, 0, 0, 0, 0, 0, 0, 0};
      __builtin_amdgcn_tensor_load_to_lds(g0, g1, z4, z4, z8, 0);
    }
    __builtin_amdgcn_s_wait_tensorcnt(0);
    __syncthreads();                        // tile visible to all waves

    float xxc = xx[b * NPT + mb + lo];      // ||x_col||^2 for my column
    v8f c8 = {0.f, 0.f, 0.f, 0.f, 0.f, 0.f, 0.f, 0.f};
    #pragma unroll
    for (int s = 0; s < 48; ++s) {
      v2f aa = {a0[s], a1[s]};
      v2f bb = {bstage[4 * s + 2 * hi][lo], bstage[4 * s + 2 * hi + 1][lo]};
      c8 = __builtin_amdgcn_wmma_f32_16x16x4_f32(false, aa, false, bb,
                                                 (short)0, c8, false, false);
    }
    // score = 2*inner - ||x_col||^2  (row-constant term dropped: same ranking)
    #pragma unroll
    for (int v = 0; v < 8; ++v)
      ctile[w][v + 8 * hi][lo] = 2.0f * c8[v] - xxc;
    __syncthreads();

    // each lane owns row (lane&15); halves split the 16 columns 8/8
    const int row = lo, cb = 8 * hi;
    #pragma unroll
    for (int j = 0; j < 8; ++j) {
      float val = ctile[w][row][cb + j];
      int   id  = mb + cb + j;
      if (val > lv[10]) {                   // rare after warm-up
        float vv = val; int vi = id;
        #pragma unroll
        for (int q = 0; q < 11; ++q) {
          bool g = vv > lv[q];
          float tv = lv[q]; int ti = li[q];
          lv[q] = g ? vv : tv;  li[q] = g ? vi : ti;
          vv    = g ? tv : vv;  vi    = g ? ti : vi;
        }
      }
    }
  }

  __syncthreads();
  #pragma unroll
  for (int j = 0; j < 11; ++j) {
    mval[w][lo][hi * 11 + j] = lv[j];
    midx[w][lo][hi * 11 + j] = li[j];
  }
  __syncthreads();
  if (hi == 0) {                            // merge two sorted half-lists
    int p = 0, q = 11;
    int* op = idx_out + (size_t)(b * NPT + nb + lo) * KNB;
    for (int j = 0; j < 11; ++j) {
      float va = (p < 11) ? mval[w][lo][p] : -3.0e38f;
      float vb = (q < 22) ? mval[w][lo][q] : -3.0e38f;
      bool ta = (p < 11) && (q >= 22 || va >= vb);
      op[j] = ta ? midx[w][lo][p] : midx[w][lo][q];
      if (ta) ++p; else ++q;
    }
  }
}

// ===========================================================================
// Kernels C/E: async-gather + dual bf16 WMMA GEMM (p = Wf*feat, d = Wd*feat).
// One workgroup per (b,n); K=11 padded to 16 cols, 3 d-slices -> 48 cols.
// Neighbor gather uses GLOBAL_LOAD_ASYNC_TO_LDS_B32 (per-lane scattered,
// tracked with ASYNCcnt) -- xf is L2-resident (12.6 MB << 192 MB L2).
// FINAL=false: accumulate per-channel sum/sumsq of ||p|| (BN stats).
// FINAL=true : apply BN + VN nonlinearity + mean over K, write out.
// ===========================================================================
template <bool FINAL>
__global__ __launch_bounds__(128) void vn_gemm_kernel(
    const float* __restrict__ x, const float* __restrict__ Wf,
    const float* __restrict__ Wd, const float* __restrict__ beta,
    const int* __restrict__ idx, const float* __restrict__ params,
    float* __restrict__ stats, float* __restrict__ out) {
  __shared__ float feat[48 * 68];           // col-major f32, stride 68 (pad)
  __shared__ int   ms[16];

  const int tid = threadIdx.x, w = tid >> 5, lane = tid & 31;
  const int lo = lane & 15, hi = lane >> 4;
  const int wg = blockIdx.x;
  const int b = wg >> 11, n = wg & (NPT - 1);

  if (tid < 16) ms[tid] = idx[(size_t)(b * NPT + n) * KNB + (tid < KNB ? tid : 0)];
  __syncthreads();

  // async gather of neighbor feature tile [cin=64][col=48] straight into LDS
  const float* xb = x + (size_t)b * DFE * NPT;
  const unsigned fbase = lds_off(&feat[0]);
  #pragma unroll
  for (int i = 0; i < 24; ++i) {
    int f = tid + 128 * i;                  // 0..3071
    int col = f >> 6, cin = f & 63;
    int d = col >> 4, t = col & 15;
    const float* gp = xb + (size_t)(cin * 3 + d) * NPT + ms[t];
    unsigned lp = fbase + (unsigned)((col * 68 + cin) * 4);
    asm volatile("global_load_async_to_lds_b32 %0, %1, off"
                 :: "v"(lp), "v"(gp) : "memory");
  }

  // A fragments for this wave's 16 output rows (bf16 16x32; k order used here, same as B: one
  // contiguous 16-long K run per lane, +32 for the second fragment)
  v16bf af0, af1, ad0, ad1;
  {
    int m = w * 16 + lo;
    const float* wr = Wf + m * CIN;
    const float* dr = Wd + m * CIN;
    #pragma unroll
    for (int j = 0; j < 16; ++j) {
      int kk = 16 * hi + j;
      af0[j] = (__bf16)wr[kk];  af1[j] = (__bf16)wr[kk + 32];
      ad0[j] = (__bf16)dr[kk];  ad1[j] = (__bf16)dr[kk + 32];
    }
  }
  asm volatile("s_wait_asynccnt 0x0" ::: "memory");
  __syncthreads();

  v8f p[3], dd[3];
  #pragma unroll
  for (int co = 0; co < 3; ++co) {          // col-tile == d component
    int colbase = co * 16 + lo;
    const float* fp = &feat[colbase * 68 + 16 * hi];
    v16bf B0v, B1v;
    #pragma unroll
    for (int j = 0; j < 16; ++j) {          // contiguous -> ds_load_b128 x4
      B0v[j] = (__bf16)fp[j];
      B1v[j] = (__bf16)fp[32 + j];
    }
    v8f z = {0.f, 0.f, 0.f, 0.f, 0.f, 0.f, 0.f, 0.f};
    v8f t0 = __builtin_amdgcn_wmma_f32_16x16x32_bf16(false, af0, false, B0v,
                                                     (short)0, z, false, false);
    p[co]  = __builtin_amdgcn_wmma_f32_16x16x32_bf16(false, af1, false, B1v,
                                                     (short)0, t0, false, false);
    v8f t1 = __builtin_amdgcn_wmma_f32_16x16x32_bf16(false, ad0, false, B0v,
                                                     (short)0, z, false, false);
    dd[co] = __builtin_amdgcn_wmma_f32_16x16x32_bf16(false, ad1, false, B1v,
                                                     (short)0, t1, false, false);
  }

  const bool act = lo < KNB;                // padded cols masked out
  #pragma unroll
  for (int v = 0; v < 8; ++v) {
    float p0 = p[0][v], p1 = p[1][v], p2 = p[2][v];
    float nrm2 = p0 * p0 + p1 * p1 + p2 * p2;
    float nrm  = sqrtf(nrm2);
    int cout = w * 16 + v + 8 * hi;
    if (!FINAL) {
      float s1 = act ? nrm : 0.f, s2 = act ? nrm2 : 0.f;
      #pragma unroll
      for (int off = 8; off >= 1; off >>= 1) {
        s1 += __shfl_xor(s1, off, 16);
        s2 += __shfl_xor(s2, off, 16);
      }
      if (lo == 0) {
        atomicAdd(&stats[cout], s1);
        atomicAdd(&stats[64 + cout], s2);
      }
    } else {
      float mean = params[cout], scale = params[64 + cout], bet = beta[cout];
      float factor = ((nrm - mean) * scale + bet) / nrm;
      float q0 = p0 * factor, q1 = p1 * factor, q2 = p2 * factor;
      float d0 = dd[0][v], d1 = dd[1][v], d2 = dd[2][v];
      float dot = q0 * d0 + q1 * d1 + q2 * d2;
      float dsq = d0 * d0 + d1 * d1 + d2 * d2;
      float adj = (dot < 0.f) ? dot / (dsq + 1e-6f) : 0.f;
      q0 -= adj * d0; q1 -= adj * d1; q2 -= adj * d2;
      if (!act) { q0 = 0.f; q1 = 0.f; q2 = 0.f; }
      #pragma unroll
      for (int off = 8; off >= 1; off >>= 1) {
        q0 += __shfl_xor(q0, off, 16);
        q1 += __shfl_xor(q1, off, 16);
        q2 += __shfl_xor(q2, off, 16);
      }
      if (lo == 0) {
        const float inv = 1.0f / (float)KNB;
        size_t base = ((size_t)(b * COUT + cout) * 3) * NPT + n;
        out[base]           = q0 * inv;
        out[base + NPT]     = q1 * inv;
        out[base + 2 * NPT] = q2 * inv;
      }
    }
  }
}

// ===========================================================================
// Kernel D: finalize BN params: mean[c], gamma[c]*rsqrt(var+eps)
// ===========================================================================
__global__ void vn_bnfin_kernel(const float* __restrict__ stats,
                                const float* __restrict__ gamma,
                                float* __restrict__ params) {
  int c = threadIdx.x;
  if (c >= COUT) return;
  const float cnt = (float)B_ * (float)NPT * (float)KNB;
  float mean = stats[c] / cnt;
  float var  = stats[64 + c] / cnt - mean * mean;
  params[c]      = mean;
  params[64 + c] = gamma[c] * rsqrtf(var + 1e-5f);
}

// ===========================================================================
extern "C" void kernel_launch(void* const* d_in, const int* in_sizes, int n_in,
                              void* d_out, int out_size, void* d_ws, size_t ws_size,
                              hipStream_t stream) {
  const float* x     = (const float*)d_in[0];
  const float* Wf    = (const float*)d_in[1];
  const float* Wd    = (const float*)d_in[2];
  const float* gamma = (const float*)d_in[3];
  const float* beta  = (const float*)d_in[4];
  // d_in[5] is k (==10, fixed shape); KNB = k+1 hardcoded.

  float* xx     = (float*)d_ws;            // B*N floats
  float* stats  = xx + B_ * NPT;           // 128 floats (sum, sumsq)
  float* params = stats + 128;             // 128 floats (mean, scale)
  int*   idx    = (int*)(params + 128);    // B*N*KNB ints
  float* out    = (float*)d_out;

  vn_xx_kernel<<<(B_ * NPT) / 256, 256, 0, stream>>>(x, xx, stats);
  vn_knn_kernel<<<(B_ * NPT) / 64, 128, 0, stream>>>(x, xx, idx);
  vn_gemm_kernel<false><<<B_ * NPT, 128, 0, stream>>>(x, Wf, Wd, beta, idx,
                                                      params, stats, out);
  vn_bnfin_kernel<<<1, 64, 0, stream>>>(stats, gamma, params);
  vn_gemm_kernel<true><<<B_ * NPT, 128, 0, stream>>>(x, Wf, Wd, beta, idx,
                                                     params, stats, out);
}
